// OptProjNN_62809601736850
// MI455X (gfx1250) — matrix operand order repaired and checked by reading, not compiled, not run on hardware
//
#include <hip/hip_runtime.h>
#include <stdint.h>

#define B_M   256
#define HID   1024
#define MDIM  2048
#define NDIM  4096
#define FREE  512
#define NITER 31          // MAX_ITER + 1 scan steps
#define EQ_TOLF   1e-4f
#define INEQ_TOLF 1e-4f
#define LN_EPSF   1e-5f
#define NOACT (1 << 30)

typedef unsigned short ush;
typedef __attribute__((ext_vector_type(16))) __bf16 v16bf;
typedef __attribute__((ext_vector_type(8)))  __bf16 v8bf;
typedef __attribute__((ext_vector_type(8)))  float  v8f;
typedef __attribute__((ext_vector_type(4)))  int    v4i;

union FragU { v16bf v; struct { v8bf lo, hi; } h; };

#if __has_builtin(__builtin_amdgcn_global_load_async_to_lds_b128)
#define ASYNC_CP 1
#else
#define ASYNC_CP 0
#endif

#if ASYNC_CP && __has_builtin(__builtin_amdgcn_s_wait_asynccnt)
#define WAIT_ASYNC(n) __builtin_amdgcn_s_wait_asynccnt(n)
#elif ASYNC_CP
#define WAIT_ASYNC(n) asm volatile("s_wait_asynccnt %0" ::"i"(n) : "memory")
#else
#define WAIT_ASYNC(n) ((void)0)
#endif

// 16-byte global -> LDS copy; async DMA (no VGPR round trip) when available.
__device__ __forceinline__ void cp16(const void* g, void* l) {
#if ASYNC_CP
  __builtin_amdgcn_global_load_async_to_lds_b128(
      (__attribute__((address_space(1))) v4i*)g,
      (__attribute__((address_space(3))) v4i*)l, 0, 0);
#else
  *(uint4*)l = *(const uint4*)g;
#endif
}

__device__ __forceinline__ ush f2bf(float f) {
  unsigned u = __float_as_uint(f);
  unsigned r = 0x7FFFu + ((u >> 16) & 1u);
  return (ush)((u + r) >> 16);
}
__device__ __forceinline__ float bf2f(ush h) {
  return __uint_as_float(((unsigned)h) << 16);
}
__device__ __forceinline__ void split2(float v, ush& h, ush& l) {
  h = f2bf(v);
  l = f2bf(v - bf2f(h));
}
__device__ __forceinline__ void pack4(const float4& v, uint2& ph, uint2& pl) {
  ush h0, h1, h2, h3, l0, l1, l2, l3;
  split2(v.x, h0, l0); split2(v.y, h1, l1);
  split2(v.z, h2, l2); split2(v.w, h3, l3);
  ph.x = (unsigned)h0 | ((unsigned)h1 << 16);
  ph.y = (unsigned)h2 | ((unsigned)h3 << 16);
  pl.x = (unsigned)l0 | ((unsigned)l1 << 16);
  pl.y = (unsigned)l2 | ((unsigned)l3 << 16);
}

// ---------------------------------------------------------------------------
// Elementwise fp32 -> (hi, lo) bf16 split, optional tail-ReLU on col>=relu_from.
__global__ __launch_bounds__(256)
void split_x_k(const float* __restrict__ X, ush* __restrict__ hi,
               ush* __restrict__ lo, int K, int relu_from)
{
  int i4 = blockIdx.x * blockDim.x + threadIdx.x;
  float4 v = ((const float4*)X)[i4];
  int col = (i4 * 4) & (K - 1);
  if (col >= relu_from) {
    v.x = fmaxf(v.x, 0.0f); v.y = fmaxf(v.y, 0.0f);
    v.z = fmaxf(v.z, 0.0f); v.w = fmaxf(v.w, 0.0f);
  }
  uint2 ph, pl;
  pack4(v, ph, pl);
  *(uint2*)&hi[(size_t)i4 * 4] = ph;
  *(uint2*)&lo[(size_t)i4 * 4] = pl;
}

// Transpose split for [K][N] row-major weights -> hi/lo stored [N][K].
__global__ __launch_bounds__(256)
void split_w_t_k(const float* __restrict__ W, ush* __restrict__ hi,
                 ush* __restrict__ lo, int N, int K)
{
  int idx = blockIdx.x * blockDim.x + threadIdx.x;   // over K*N, coalesced read
  int n = idx & (N - 1), k = idx / N;
  float v = W[idx];
  ush h, l;
  split2(v, h, l);
  hi[(size_t)n * K + k] = h;
  lo[(size_t)n * K + k] = l;
}

// ---------------------------------------------------------------------------
// C[M,N] = epilogue( X @ W + bias ), X pre-split bf16 hi/lo [M][K].
// If Whi!=null : W pre-split bf16 [N][K]  -> pure-copy staging, async double-buffered.
// Else         : W f32; w_nk=1 -> [N][K] layout, w_nk=0 -> [K][N] (transpose staging).
// Optional fused output split: Shi/Slo get split(ReLU-tail(C)) with col>=srelu_from.
// Split-bf16 fp32 emulation: acc += Ahi*Bhi + Ahi*Blo + Alo*Bhi (f32 accum).
// Block tile 128x64, 8 waves, each wave 32x32 -> 12 WMMA per wave per K=32 step.
__global__ __launch_bounds__(256)
void gemm_bf16x3(const ush* __restrict__ Xhi, const ush* __restrict__ Xlo,
                 const ush* __restrict__ Whi, const ush* __restrict__ Wlo,
                 const float* __restrict__ W,
                 float* __restrict__ C,
                 const float* __restrict__ bias_vec,   // [N] or null
                 const float* __restrict__ bias_mat,   // [M,N] or null
                 ush* __restrict__ Shi, ush* __restrict__ Slo,  // split out or null
                 int M, int N, int K, int w_nk, int relu_out, int srelu_from)
{
  __shared__ __align__(16) ush a_hi[2][128][32], a_lo[2][128][32];
  __shared__ __align__(16) ush b_hi[2][64][32],  b_lo[2][64][32];

  const int tid  = threadIdx.x;
  const int lane = tid & 31;
  const int wid  = tid >> 5;
  const int wm   = wid & 3;     // M sub-block: 32 rows
  const int wn   = wid >> 2;    // N half: 32 cols
  const int m0   = blockIdx.x * 128;
  const int n0   = blockIdx.y * 64;
  const int mrow = lane & 15;
  const int kh   = lane >> 4;

  v8f zero = 0.0f;
  v8f acc[2][2];
  acc[0][0] = zero; acc[0][1] = zero; acc[1][0] = zero; acc[1][1] = zero;

  if (Whi) {
    // ---------------- pure-copy staging, double-buffered ----------------
    auto stage = [&](int buf, int kc) {
#pragma unroll
      for (int i = 0; i < 2; ++i) {
        int f = tid + i * 256;          // ushort8 index in 128x4 grid
        int r = f >> 2, k = (f & 3) * 8;
        size_t g = (size_t)(m0 + r) * K + kc + k;
        cp16(&Xhi[g], &a_hi[buf][r][k]);
        cp16(&Xlo[g], &a_lo[buf][r][k]);
      }
      {
        int r = tid >> 2, k = (tid & 3) * 8;   // 64x4 grid
        size_t g = (size_t)(n0 + r) * K + kc + k;
        cp16(&Whi[g], &b_hi[buf][r][k]);
        cp16(&Wlo[g], &b_lo[buf][r][k]);
      }
    };
    stage(0, 0);
    int cur = 0;
    for (int kc = 0; kc < K; kc += 32) {
      const bool more = (kc + 32) < K;
      if (more) stage(cur ^ 1, kc + 32);   // prefetch next tile
      if (more) { WAIT_ASYNC(6); } else { WAIT_ASYNC(0); }
      __syncthreads();                      // current tile visible

      FragU Ah[2], Al[2];
#pragma unroll
      for (int mt = 0; mt < 2; ++mt) {
        int ar = wm * 32 + mt * 16 + mrow;
        Ah[mt].h.lo = *(const v8bf*)&a_hi[cur][ar][kh * 8];
        Ah[mt].h.hi = *(const v8bf*)&a_hi[cur][ar][16 + kh * 8];
        Al[mt].h.lo = *(const v8bf*)&a_lo[cur][ar][kh * 8];
        Al[mt].h.hi = *(const v8bf*)&a_lo[cur][ar][16 + kh * 8];
      }
#pragma unroll
      for (int nt = 0; nt < 2; ++nt) {
        int nb = wn * 32 + nt * 16 + mrow;
        FragU Bh, Bl;
        Bh.h.lo = *(const v8bf*)&b_hi[cur][nb][kh * 8];
        Bh.h.hi = *(const v8bf*)&b_hi[cur][nb][16 + kh * 8];
        Bl.h.lo = *(const v8bf*)&b_lo[cur][nb][kh * 8];
        Bl.h.hi = *(const v8bf*)&b_lo[cur][nb][16 + kh * 8];
#pragma unroll
        for (int mt = 0; mt < 2; ++mt) {
          acc[mt][nt] = __builtin_amdgcn_wmma_f32_16x16x32_bf16(
              false, Ah[mt].v, false, Bh.v, (short)0, acc[mt][nt], false, false);
          acc[mt][nt] = __builtin_amdgcn_wmma_f32_16x16x32_bf16(
              false, Ah[mt].v, false, Bl.v, (short)0, acc[mt][nt], false, false);
          acc[mt][nt] = __builtin_amdgcn_wmma_f32_16x16x32_bf16(
              false, Al[mt].v, false, Bh.v, (short)0, acc[mt][nt], false, false);
        }
      }
      __syncthreads();                      // reads of cur done -> reusable
      cur ^= 1;
    }
  } else {
    // ---------------- convert-W staging (single buffer) ----------------
    for (int kc = 0; kc < K; kc += 32) {
#pragma unroll
      for (int i = 0; i < 2; ++i) {
        int f = tid + i * 256;
        int r = f >> 2, k = (f & 3) * 8;
        size_t g = (size_t)(m0 + r) * K + kc + k;
        cp16(&Xhi[g], &a_hi[0][r][k]);
        cp16(&Xlo[g], &a_lo[0][r][k]);
      }
      if (w_nk) {
        int r0 = tid >> 3, k0 = (tid & 7) * 4;
        int r1 = (tid + 256) >> 3, k1 = ((tid + 256) & 7) * 4;
        float4 v0 = *(const float4*)&W[(size_t)(n0 + r0) * K + kc + k0];
        float4 v1 = *(const float4*)&W[(size_t)(n0 + r1) * K + kc + k1];
        uint2 ph, pl;
        pack4(v0, ph, pl);
        *(uint2*)&b_hi[0][r0][k0] = ph; *(uint2*)&b_lo[0][r0][k0] = pl;
        pack4(v1, ph, pl);
        *(uint2*)&b_hi[0][r1][k1] = ph; *(uint2*)&b_lo[0][r1][k1] = pl;
      } else {
        float vv[8];
#pragma unroll
        for (int i = 0; i < 8; ++i) {
          int e = tid + i * 256;
          vv[i] = W[(size_t)(kc + (e >> 6)) * N + n0 + (e & 63)];
        }
#pragma unroll
        for (int i = 0; i < 8; ++i) {
          int e = tid + i * 256;
          ush h, l;
          split2(vv[i], h, l);
          b_hi[0][e & 63][e >> 6] = h;
          b_lo[0][e & 63][e >> 6] = l;
        }
      }
      WAIT_ASYNC(0);
      __syncthreads();

      FragU Ah[2], Al[2];
#pragma unroll
      for (int mt = 0; mt < 2; ++mt) {
        int ar = wm * 32 + mt * 16 + mrow;
        Ah[mt].h.lo = *(const v8bf*)&a_hi[0][ar][kh * 8];
        Ah[mt].h.hi = *(const v8bf*)&a_hi[0][ar][16 + kh * 8];
        Al[mt].h.lo = *(const v8bf*)&a_lo[0][ar][kh * 8];
        Al[mt].h.hi = *(const v8bf*)&a_lo[0][ar][16 + kh * 8];
      }
#pragma unroll
      for (int nt = 0; nt < 2; ++nt) {
        int nb = wn * 32 + nt * 16 + mrow;
        FragU Bh, Bl;
        Bh.h.lo = *(const v8bf*)&b_hi[0][nb][kh * 8];
        Bh.h.hi = *(const v8bf*)&b_hi[0][nb][16 + kh * 8];
        Bl.h.lo = *(const v8bf*)&b_lo[0][nb][kh * 8];
        Bl.h.hi = *(const v8bf*)&b_lo[0][nb][16 + kh * 8];
#pragma unroll
        for (int mt = 0; mt < 2; ++mt) {
          acc[mt][nt] = __builtin_amdgcn_wmma_f32_16x16x32_bf16(
              false, Ah[mt].v, false, Bh.v, (short)0, acc[mt][nt], false, false);
          acc[mt][nt] = __builtin_amdgcn_wmma_f32_16x16x32_bf16(
              false, Ah[mt].v, false, Bl.v, (short)0, acc[mt][nt], false, false);
          acc[mt][nt] = __builtin_amdgcn_wmma_f32_16x16x32_bf16(
              false, Al[mt].v, false, Bh.v, (short)0, acc[mt][nt], false, false);
        }
      }
      __syncthreads();
    }
  }

  // ---- epilogue: C 16x16 layout: VGPR i -> M = i + (lane>>4)*8, N = lane&15 ----
#pragma unroll
  for (int mt = 0; mt < 2; ++mt)
#pragma unroll
    for (int nt = 0; nt < 2; ++nt) {
      v8f a = acc[mt][nt];
      int cn = n0 + wn * 32 + nt * 16 + (lane & 15);
      int rbase = m0 + wm * 32 + mt * 16 + (lane >> 4) * 8;
      float bv = bias_vec ? bias_vec[cn] : 0.0f;
#pragma unroll
      for (int i = 0; i < 8; ++i) {
        int rm = rbase + i;
        float v = a[i] + bv;
        if (bias_mat) v += bias_mat[(size_t)rm * N + cn];
        if (relu_out) v = fmaxf(v, 0.0f);
        C[(size_t)rm * N + cn] = v;
        if (Shi) {   // fused split of the output (with optional tail-ReLU view)
          float sv = (cn >= srelu_from) ? fmaxf(v, 0.0f) : v;
          ush h, l;
          split2(sv, h, l);
          Shi[(size_t)rm * N + cn] = h;
          Slo[(size_t)rm * N + cn] = l;
        }
      }
    }
}

// ---------------------------------------------------------------------------
// Row LayerNorm, H=1024, one block per row, in-place; emits bf16 hi/lo split.
__global__ __launch_bounds__(256)
void layernorm_k(float* __restrict__ x, const float* __restrict__ g,
                 const float* __restrict__ bb,
                 ush* __restrict__ shi, ush* __restrict__ slo)
{
  const int row = blockIdx.x;
  float* xr = x + (size_t)row * HID;
  const int tid = threadIdx.x;
  float vals[4];
  float s = 0.0f;
#pragma unroll
  for (int i = 0; i < 4; ++i) { vals[i] = xr[tid + i * 256]; s += vals[i]; }
  __shared__ float red[8];
  for (int off = 16; off; off >>= 1) s += __shfl_xor(s, off, 32);
  if ((tid & 31) == 0) red[tid >> 5] = s;
  __syncthreads();
  float tot = 0.0f;
#pragma unroll
  for (int w = 0; w < 8; ++w) tot += red[w];
  const float mu = tot * (1.0f / HID);
  float vs = 0.0f;
#pragma unroll
  for (int i = 0; i < 4; ++i) { float d = vals[i] - mu; vs += d * d; }
  for (int off = 16; off; off >>= 1) vs += __shfl_xor(vs, off, 32);
  __syncthreads();
  if ((tid & 31) == 0) red[tid >> 5] = vs;
  __syncthreads();
  float vtot = 0.0f;
#pragma unroll
  for (int w = 0; w < 8; ++w) vtot += red[w];
  const float inv = rsqrtf(vtot * (1.0f / HID) + LN_EPSF);
#pragma unroll
  for (int i = 0; i < 4; ++i) {
    int c = tid + i * 256;
    float y = (vals[i] - mu) * inv * g[c] + bb[c];
    xr[c] = y;
    ush h, l;
    split2(y, h, l);
    shi[(size_t)row * HID + c] = h;
    slo[(size_t)row * HID + c] = l;
  }
}

// eq_j = mean_b |E[b,j] - bp[b,j]|  > tol  -> viol
__global__ void reduce_eq_k(const float* __restrict__ E,
                            const float* __restrict__ bp, int* __restrict__ flags)
{
  int j = blockIdx.x * blockDim.x + threadIdx.x;
  float s = 0.0f;
  for (int b = 0; b < B_M; ++b) s += fabsf(E[b * MDIM + j] - bp[b * MDIM + j]);
  if (s > EQ_TOLF * (float)B_M) atomicOr(&flags[2], 1);
}

// ineq_j = mean_b relu(-zn[b, 512+j]) > tol -> viol
__global__ void reduce_ineq_k(const float* __restrict__ zn, int* __restrict__ flags)
{
  int j = FREE + blockIdx.x * blockDim.x + threadIdx.x;
  float s = 0.0f;
  for (int b = 0; b < B_M; ++b) s += fmaxf(-zn[b * NDIM + j], 0.0f);
  if (s > INEQ_TOLF * (float)B_M) atomicOr(&flags[2], 1);
}

__global__ void init_flags_k(int* flags) { flags[0] = 0; flags[1] = 0; flags[2] = 0; }

// If !done (read BEFORE update_flags_k): z = zn and pz = split(P2(zn)).
__global__ __launch_bounds__(256)
void commit_z_k(float* __restrict__ z, const float* __restrict__ zn,
                ush* __restrict__ pz_hi, ush* __restrict__ pz_lo,
                const int* __restrict__ flags)
{
  if (flags[0]) return;   // uniform branch: frozen state already consistent
  int i4 = blockIdx.x * blockDim.x + threadIdx.x;
  float4 v = ((const float4*)zn)[i4];
  ((float4*)z)[i4] = v;
  int col = (i4 * 4) & (NDIM - 1);
  if (col >= FREE) {
    v.x = fmaxf(v.x, 0.0f); v.y = fmaxf(v.y, 0.0f);
    v.z = fmaxf(v.z, 0.0f); v.w = fmaxf(v.w, 0.0f);
  }
  uint2 ph, pl;
  pack4(v, ph, pl);
  *(uint2*)&pz_hi[(size_t)i4 * 4] = ph;
  *(uint2*)&pz_lo[(size_t)i4 * 4] = pl;
}

// it = done ? it : it+1 ; done |= stop (stop = !viol); reset viol for next iter.
__global__ void update_flags_k(int* flags)
{
  int d = flags[0];
  if (!d) flags[1] += 1;
  flags[0] = d | (flags[2] == 0 ? 1 : 0);
  flags[2] = 0;
}

__global__ void write_pn_k(const int* flags, int* out) { *out = flags[1]; }

// ---------------------------------------------------------------------------
extern "C" void kernel_launch(void* const* d_in, const int* in_sizes, int n_in,
                              void* d_out, int out_size, void* d_ws, size_t ws_size,
                              hipStream_t stream) {
  (void)in_sizes; (void)n_in; (void)out_size;
  const float* bp  = (const float*)d_in[0];
  const float* w1  = (const float*)d_in[1];
  const float* b1  = (const float*)d_in[2];
  const float* g1  = (const float*)d_in[3];
  const float* e1  = (const float*)d_in[4];
  const float* w2  = (const float*)d_in[5];
  const float* b2  = (const float*)d_in[6];
  const float* g2  = (const float*)d_in[7];
  const float* e2  = (const float*)d_in[8];
  const float* w3  = (const float*)d_in[9];
  const float* b3  = (const float*)d_in[10];
  const float* g3  = (const float*)d_in[11];
  const float* e3  = (const float*)d_in[12];
  const float* wo  = (const float*)d_in[13];
  const float* bo  = (const float*)d_in[14];
  const float* A   = (const float*)d_in[15];
  const float* Wz  = (const float*)d_in[16];  // symmetric: Wp = WzProj.T = WzProj
  const float* Wb  = (const float*)d_in[17];

  float* out = (float*)d_out;
  float* z   = out;                               // z_star [256,4096]
  float* z0  = out + (size_t)B_M * NDIM;          // z0     [256,4096]
  int*   pn  = (int*)(out + 2 * (size_t)B_M * NDIM);

  char* w = (char*)d_ws;
  auto take = [&](size_t bytes) { char* p = w; w += bytes; return p; };
  float* h1   = (float*)take((size_t)B_M * HID  * 4);
  float* h2   = (float*)take((size_t)B_M * HID  * 4);
  float* h3   = (float*)take((size_t)B_M * HID  * 4);
  float* Bias = (float*)take((size_t)B_M * NDIM * 4);
  float* zn   = (float*)take((size_t)B_M * NDIM * 4);
  float* E    = (float*)take((size_t)B_M * MDIM * 4);
  ush* bp_hi  = (ush*)take((size_t)B_M * MDIM * 2);
  ush* bp_lo  = (ush*)take((size_t)B_M * MDIM * 2);
  ush* h_hi   = (ush*)take((size_t)B_M * HID  * 2);
  ush* h_lo   = (ush*)take((size_t)B_M * HID  * 2);
  ush* z0_hi  = (ush*)take((size_t)B_M * NDIM * 2);
  ush* z0_lo  = (ush*)take((size_t)B_M * NDIM * 2);
  ush* pz_hi  = (ush*)take((size_t)B_M * NDIM * 2);
  ush* pz_lo  = (ush*)take((size_t)B_M * NDIM * 2);
  ush* zn_hi  = (ush*)take((size_t)B_M * NDIM * 2);
  ush* zn_lo  = (ush*)take((size_t)B_M * NDIM * 2);
  int* flags  = (int*)take(256);

  // --- deterministic weight pre-split tiers based on ws_size ---
  const size_t used_act = (size_t)(w - (char*)d_ws);
  const size_t rem = ws_size > used_act ? ws_size - used_act : 0;
  const size_t WZB = (size_t)NDIM * NDIM * 4;   // hi+lo bf16 bytes
  const size_t AB  = (size_t)MDIM * NDIM * 4;
  const size_t WBB = (size_t)NDIM * MDIM * 4;
  const size_t W1B = (size_t)MDIM * HID * 4;
  const size_t W2B = (size_t)HID * HID * 4;
  const size_t WOB = (size_t)HID * NDIM * 4;
  const bool t1 = rem >= (WZB + AB);                                  // Wz + A
  const bool t2 = rem >= (WZB + AB + WBB + W1B + 2 * W2B + WOB);      // all
  ush *wz_hi = nullptr, *wz_lo = nullptr, *a_hi = nullptr, *a_lo = nullptr;
  ush *wb_hi = nullptr, *wb_lo = nullptr;
  ush *w1_hi = nullptr, *w1_lo = nullptr, *w2_hi = nullptr, *w2_lo = nullptr;
  ush *w3_hi = nullptr, *w3_lo = nullptr, *wo_hi = nullptr, *wo_lo = nullptr;
  if (t1) {
    wz_hi = (ush*)take(WZB / 2); wz_lo = (ush*)take(WZB / 2);
    a_hi  = (ush*)take(AB / 2);  a_lo  = (ush*)take(AB / 2);
  }
  if (t2) {
    wb_hi = (ush*)take(WBB / 2); wb_lo = (ush*)take(WBB / 2);
    w1_hi = (ush*)take(W1B / 2); w1_lo = (ush*)take(W1B / 2);
    w2_hi = (ush*)take(W2B / 2); w2_lo = (ush*)take(W2B / 2);
    w3_hi = (ush*)take(W2B / 2); w3_lo = (ush*)take(W2B / 2);
    wo_hi = (ush*)take(WOB / 2); wo_lo = (ush*)take(WOB / 2);
  }

  const dim3 blk(256);
  const int G_BP = (B_M * MDIM / 4) / 256;   // 512
  const int G_Z  = (B_M * NDIM / 4) / 256;   // 1024

  if (t1) {
    split_x_k<<<(NDIM * NDIM / 4) / 256, blk, 0, stream>>>(Wz, wz_hi, wz_lo, NDIM, NOACT);
    split_x_k<<<(MDIM * NDIM / 4) / 256, blk, 0, stream>>>(A, a_hi, a_lo, NDIM, NOACT);
  }
  if (t2) {
    split_x_k<<<(NDIM * MDIM / 4) / 256, blk, 0, stream>>>(Wb, wb_hi, wb_lo, MDIM, NOACT);
    split_w_t_k<<<(MDIM * HID) / 256, blk, 0, stream>>>(w1, w1_hi, w1_lo, HID, MDIM);
    split_w_t_k<<<(HID * HID) / 256, blk, 0, stream>>>(w2, w2_hi, w2_lo, HID, HID);
    split_w_t_k<<<(HID * HID) / 256, blk, 0, stream>>>(w3, w3_hi, w3_lo, HID, HID);
    split_w_t_k<<<(HID * NDIM) / 256, blk, 0, stream>>>(wo, wo_hi, wo_lo, NDIM, HID);
  }

  // ---- MLP (LayerNorm emits split activations; GEMMs emit split outputs) ----
  split_x_k<<<G_BP, blk, 0, stream>>>(bp, bp_hi, bp_lo, MDIM, NOACT);
  gemm_bf16x3<<<dim3(2, 16), blk, 0, stream>>>(bp_hi, bp_lo, w1_hi, w1_lo, w1, h1,
                                               b1, nullptr, nullptr, nullptr,
                                               B_M, HID, MDIM, 0, 1, NOACT);
  layernorm_k<<<B_M, blk, 0, stream>>>(h1, g1, e1, h_hi, h_lo);
  gemm_bf16x3<<<dim3(2, 16), blk, 0, stream>>>(h_hi, h_lo, w2_hi, w2_lo, w2, h2,
                                               b2, nullptr, nullptr, nullptr,
                                               B_M, HID, HID, 0, 1, NOACT);
  layernorm_k<<<B_M, blk, 0, stream>>>(h2, g2, e2, h_hi, h_lo);
  gemm_bf16x3<<<dim3(2, 16), blk, 0, stream>>>(h_hi, h_lo, w3_hi, w3_lo, w3, h3,
                                               b3, nullptr, nullptr, nullptr,
                                               B_M, HID, HID, 0, 1, NOACT);
  layernorm_k<<<B_M, blk, 0, stream>>>(h3, g3, e3, h_hi, h_lo);
  gemm_bf16x3<<<dim3(2, 64), blk, 0, stream>>>(h_hi, h_lo, wo_hi, wo_lo, wo, z0,
                                               bo, nullptr, z0_hi, z0_lo,
                                               B_M, NDIM, HID, 0, 0, NOACT);

  // ---- Bias = bp @ WbProj.T ----
  gemm_bf16x3<<<dim3(2, 64), blk, 0, stream>>>(bp_hi, bp_lo, wb_hi, wb_lo, Wb, Bias,
                                               nullptr, nullptr, nullptr, nullptr,
                                               B_M, NDIM, MDIM, 1, 0, NOACT);
  // ---- z_init = Bias + z0 @ Wp ; emits first pz = split(P2(z_init)) ----
  gemm_bf16x3<<<dim3(2, 64), blk, 0, stream>>>(z0_hi, z0_lo, wz_hi, wz_lo, Wz, z,
                                               nullptr, Bias, pz_hi, pz_lo,
                                               B_M, NDIM, NDIM, 1, 0, FREE);

  init_flags_k<<<1, 1, 0, stream>>>(flags);

  for (int it = 0; it < NITER; ++it) {
    // zn = Bias + P2(z) @ Wp ; emits zn split for the E GEMM
    gemm_bf16x3<<<dim3(2, 64), blk, 0, stream>>>(pz_hi, pz_lo, wz_hi, wz_lo, Wz, zn,
                                                 nullptr, Bias, zn_hi, zn_lo,
                                                 B_M, NDIM, NDIM, 1, 0, NOACT);
    // E = zn @ A.T
    gemm_bf16x3<<<dim3(2, 32), blk, 0, stream>>>(zn_hi, zn_lo, a_hi, a_lo, A, E,
                                                 nullptr, nullptr, nullptr, nullptr,
                                                 B_M, MDIM, NDIM, 1, 0, NOACT);
    reduce_eq_k<<<MDIM / 256, blk, 0, stream>>>(E, bp, flags);
    reduce_ineq_k<<<(NDIM - FREE) / 256, blk, 0, stream>>>(zn, flags);
    commit_z_k<<<G_Z, blk, 0, stream>>>(z, zn, pz_hi, pz_lo, flags);
    update_flags_k<<<1, 1, 0, stream>>>(flags);
  }
  write_pn_k<<<1, 1, 0, stream>>>(flags, pn);
}
